// MultiheadSelfAttention_16372415332420
// MI455X (gfx1250) — matrix operand order repaired and checked by reading, not compiled, not run on hardware
//
#include <hip/hip_runtime.h>
#include <hip/hip_bf16.h>

typedef _Float16 f16;
typedef __attribute__((ext_vector_type(16))) _Float16 v16h;
typedef __attribute__((ext_vector_type(8)))  _Float16 v8h;
typedef __attribute__((ext_vector_type(2)))  _Float16 h2;
typedef __attribute__((ext_vector_type(8)))  float    v8f;

#define Tn   2048
#define Bn   4
#define En   1024
#define Hn   16
#define HDn  64
#define NQKV 3072
#define MROWS (Tn*Bn)   // 8192

#if __has_builtin(__builtin_amdgcn_global_load_async_to_lds_b128) && \
    __has_builtin(__builtin_amdgcn_s_wait_asynccnt)
#define ASYNC_LDS 1
typedef __attribute__((ext_vector_type(4))) int i4;
typedef __attribute__((address_space(1))) i4* as1_i4p;
typedef __attribute__((address_space(3))) i4* as3_i4p;
#endif

static __device__ __forceinline__ v8f wmma_f16(v16h a, v16h b, v8f c) {
    return __builtin_amdgcn_wmma_f32_16x16x32_f16(false, a, false, b, (short)0, c, false, false);
}

// ---------------------------------------------------------------------------
// Kernel 1: fused QKV projection.  proj = X(8192,1024) @ Win(3072,1024)^T + bin
// Software-pipelined: next k-tile is loaded to registers while WMMA runs.
// Epilogue scatters into Q/K/V buffers [B*H][T][HD] f16, Q scaled by 0.125.
// ---------------------------------------------------------------------------
__launch_bounds__(256)
__global__ void qkv_proj_kernel(const float* __restrict__ X,
                                const float* __restrict__ Win,
                                const float* __restrict__ bin,
                                f16* __restrict__ Qb, f16* __restrict__ Kb,
                                f16* __restrict__ Vb)
{
    constexpr int BM = 128, BN = 128, BK = 32;
    constexpr int AST = 34;    // As row stride (halves)
    constexpr int BST = 136;   // Bs row stride (halves), 272B = 17*16
    __shared__ f16 As[BM * AST];   // [m][k]
    __shared__ f16 Bs[BK * BST];   // [k][n]  (W transposed)

    const int tid  = threadIdx.x;
    const int lane = tid & 31, wave = tid >> 5;
    const int wm = wave >> 1, wn = wave & 1;      // 4x2 wave grid
    const int g = lane >> 4, lm = lane & 15;
    const int m0 = blockIdx.x * BM;
    const int n0 = blockIdx.y * BN;

    v8f zero8 = {0.f,0.f,0.f,0.f,0.f,0.f,0.f,0.f};
    v8f acc[2][4];
#pragma unroll
    for (int mi = 0; mi < 2; ++mi)
#pragma unroll
        for (int ni = 0; ni < 4; ++ni) acc[mi][ni] = zero8;

    const int ar = tid >> 3;            // 0..31
    const int ac = (tid & 7) * 4;       // 0..28

    float4 aT[4], wT[4];
    auto load_tiles = [&](int k0) {
#pragma unroll
        for (int p = 0; p < 4; ++p)
            aT[p] = *(const float4*)&X[(size_t)(m0 + p * 32 + ar) * En + k0 + ac];
#pragma unroll
        for (int p = 0; p < 4; ++p)
            wT[p] = *(const float4*)&Win[(size_t)(n0 + p * 32 + ar) * En + k0 + ac];
    };
    load_tiles(0);

    for (int k0 = 0; k0 < En; k0 += BK) {
        __syncthreads();
        // stage prefetched registers to LDS (f32 -> f16)
#pragma unroll
        for (int p = 0; p < 4; ++p) {
            int row = p * 32 + ar;
            h2 p0 = {(f16)aT[p].x, (f16)aT[p].y};
            h2 p1 = {(f16)aT[p].z, (f16)aT[p].w};
            *(h2*)&As[row * AST + ac]     = p0;
            *(h2*)&As[row * AST + ac + 2] = p1;
            int n = p * 32 + ar;
            Bs[(ac + 0) * BST + n] = (f16)wT[p].x;
            Bs[(ac + 1) * BST + n] = (f16)wT[p].y;
            Bs[(ac + 2) * BST + n] = (f16)wT[p].z;
            Bs[(ac + 3) * BST + n] = (f16)wT[p].w;
        }
        __syncthreads();
        if (k0 + BK < En) load_tiles(k0 + BK);   // in flight during WMMA below

        // A fragments (16x32): lane L: M=L%16; halves h -> K = (h/8)*16 + g*8 + h%8
        v16h af[2];
#pragma unroll
        for (int mi = 0; mi < 2; ++mi) {
            const f16* base = &As[(wm * 32 + mi * 16 + lm) * AST];
#pragma unroll
            for (int h = 0; h < 16; h += 2) {
                int k = (h >> 3) * 16 + g * 8 + (h & 7);
                h2 t = *(const h2*)&base[k];
                af[mi][h] = t[0]; af[mi][h + 1] = t[1];
            }
        }
        // B fragments (32x16): lane L: N=L%16; halves h -> K = (h/8)*16 + g*8 + h%8
#pragma unroll
        for (int ni = 0; ni < 4; ++ni) {
            const f16* bb = &Bs[wn * 64 + ni * 16 + lm];
            v16h bf;
#pragma unroll
            for (int h = 0; h < 16; ++h) bf[h] = bb[((h >> 3) * 16 + g * 8 + (h & 7)) * BST];
#pragma unroll
            for (int mi = 0; mi < 2; ++mi)
                acc[mi][ni] = wmma_f16(af[mi], bf, acc[mi][ni]);
        }
    }

    // epilogue: C layout element (M = j + 8g, N = lm)
#pragma unroll
    for (int ni = 0; ni < 4; ++ni) {
        int f = n0 + wn * 64 + ni * 16 + lm;
        float bias  = bin[f];
        int which   = f >> 10;           // 0:q 1:k 2:v
        int e       = f & 1023;
        int hh      = e >> 6;
        int hd      = e & 63;
        f16* dst    = (which == 0) ? Qb : (which == 1) ? Kb : Vb;
        float scal  = (which == 0) ? 0.125f : 1.0f;
#pragma unroll
        for (int mi = 0; mi < 2; ++mi) {
            int mbase = m0 + wm * 32 + mi * 16 + g * 8;
#pragma unroll
            for (int j = 0; j < 8; ++j) {
                int m = mbase + j;
                int t = m >> 2, b = m & 3;          // Bn = 4
                float val = (acc[mi][ni][j] + bias) * scal;
                dst[((size_t)(b * Hn + hh) * Tn + t) * HDn + hd] = (f16)val;
            }
        }
    }
}

// ---------------------------------------------------------------------------
// Kernel 2: flash attention per (b,h).  Q,K,V: [B*H][T][HD] f16.
// Block: 128 Q rows (16 per wave), loop over S in chunks of 64.
// K/V chunk is register-prefetched one chunk ahead; mask prefetched.
// Writes attn output (T*B, E) as f16.
// ---------------------------------------------------------------------------
__launch_bounds__(256)
__global__ void attn_kernel(const f16* __restrict__ Qb, const f16* __restrict__ Kb,
                            const f16* __restrict__ Vb,
                            const float* __restrict__ mask,
                            const unsigned char* __restrict__ kpm,
                            f16* __restrict__ attnout)
{
    constexpr int BS = 64;
    constexpr int KTS = 72, VSS = 72, PSS = 72;   // LDS strides (halves)
    __shared__ f16 Kt[HDn * KTS];          // [hd][s]   (transposed)
    __shared__ f16 Vs[BS * VSS];           // [s][hd]
    __shared__ f16 Ps[8 * 16 * PSS];       // per-wave 16 x 64 P scratch

    const int tid  = threadIdx.x;
    const int lane = tid & 31, wave = tid >> 5;
    const int g = lane >> 4, lm = lane & 15;
    const int bh = blockIdx.y;
    const int b  = bh >> 4;                 // /H
    const int q0 = blockIdx.x * 128 + wave * 16;

    const f16* Qh = Qb + (size_t)bh * Tn * HDn;
    const f16* Kh = Kb + (size_t)bh * Tn * HDn;
    const f16* Vh = Vb + (size_t)bh * Tn * HDn;

    // Q fragments, kept for the whole S loop (HD=64 -> 2 k-steps of 32)
    v16h aq[2];
#pragma unroll
    for (int kk = 0; kk < 2; ++kk) {
        const f16* qr = &Qh[(size_t)(q0 + lm) * HDn];
#pragma unroll
        for (int h = 0; h < 16; h += 2) {
            int hd = kk * 32 + (h >> 3) * 16 + g * 8 + (h & 7);
            h2 t = *(const h2*)&qr[hd];
            aq[kk][h] = t[0]; aq[kk][h + 1] = t[1];
        }
    }

    v8f zero8 = {0.f,0.f,0.f,0.f,0.f,0.f,0.f,0.f};
    v8f co[4];
#pragma unroll
    for (int vt = 0; vt < 4; ++vt) co[vt] = zero8;
    float mrun[8], lrun[8];
#pragma unroll
    for (int j = 0; j < 8; ++j) { mrun[j] = -3.0e38f; lrun[j] = 0.f; }

    const int sl  = tid & 63;           // staging: s index
    const int hd0 = (tid >> 6) * 16;    // staging: hd group

    // register prefetch of K/V chunk
    v8h kr0, kr1, vr0, vr1;
    auto load_chunk = [&](int s0) {
        const f16* kr = &Kh[(size_t)(s0 + sl) * HDn + hd0];
        kr0 = *(const v8h*)&kr[0];
        kr1 = *(const v8h*)&kr[8];
        const f16* vr = &Vh[(size_t)(s0 + sl) * HDn + hd0];
        vr0 = *(const v8h*)&vr[0];
        vr1 = *(const v8h*)&vr[8];
    };
    load_chunk(0);

    for (int s0 = 0; s0 < Tn; s0 += BS) {
        __syncthreads();
        // stage prefetched K (transposed) + V (row-major)
#pragma unroll
        for (int i = 0; i < 8; ++i) {
            Kt[(hd0 + i) * KTS + sl]     = kr0[i];
            Kt[(hd0 + 8 + i) * KTS + sl] = kr1[i];
        }
        *(v8h*)&Vs[sl * VSS + hd0]     = vr0;
        *(v8h*)&Vs[sl * VSS + hd0 + 8] = vr1;
        __syncthreads();
        if (s0 + BS < Tn) {
            load_chunk(s0 + BS);    // in flight during compute below
            __builtin_prefetch(&mask[(size_t)(q0 + g * 8) * Tn + s0 + BS + lm], 0, 0);
        }

        // scores: S_tile(16 x 64) = Q(16x64) @ K^T
        v8f cs[4];
#pragma unroll
        for (int st = 0; st < 4; ++st) cs[st] = zero8;
#pragma unroll
        for (int st = 0; st < 4; ++st) {
#pragma unroll
            for (int kk = 0; kk < 2; ++kk) {
                const f16* kb = &Kt[(kk * 32) * KTS + st * 16 + lm];
                v16h bf;
#pragma unroll
                for (int h = 0; h < 16; ++h) bf[h] = kb[((h >> 3) * 16 + g * 8 + (h & 7)) * KTS];
                cs[st] = wmma_f16(aq[kk], bf, cs[st]);
            }
        }

        // mask + online softmax, row by row (row M = j + 8g, col N = lm)
        f16* pw = &Ps[wave * 16 * PSS];
#pragma unroll
        for (int j = 0; j < 8; ++j) {
            int t = q0 + g * 8 + j;
            float sc[4];
            float mx = -3.0e38f;
#pragma unroll
            for (int st = 0; st < 4; ++st) {
                int s = s0 + st * 16 + lm;
                float v = cs[st][j] + mask[(size_t)t * Tn + s];
                if (kpm[b * Tn + s]) v = -1.0e9f;
                sc[st] = v;
                mx = fmaxf(mx, v);
            }
#pragma unroll
            for (int off = 8; off; off >>= 1) mx = fmaxf(mx, __shfl_xor(mx, off, 32));
            float mnew  = fmaxf(mrun[j], mx);
            float alpha = __expf(mrun[j] - mnew);
            float rsum  = 0.f;
#pragma unroll
            for (int st = 0; st < 4; ++st) {
                float pv = __expf(sc[st] - mnew);
                rsum += pv;
                pw[(g * 8 + j) * PSS + st * 16 + lm] = (f16)pv;
            }
#pragma unroll
            for (int off = 8; off; off >>= 1) rsum += __shfl_xor(rsum, off, 32);
            lrun[j] = lrun[j] * alpha + rsum;
            mrun[j] = mnew;
#pragma unroll
            for (int vt = 0; vt < 4; ++vt) co[vt][j] *= alpha;
        }

        // P (C-layout) -> A-fragments via per-wave LDS scratch (wave-local)
        v16h ap[2];
#pragma unroll
        for (int kk = 0; kk < 2; ++kk) {
            const f16* pr = &pw[lm * PSS];
#pragma unroll
            for (int h = 0; h < 16; h += 2) {
                int s = kk * 32 + (h >> 3) * 16 + g * 8 + (h & 7);
                h2 t = *(const h2*)&pr[s];
                ap[kk][h] = t[0]; ap[kk][h + 1] = t[1];
            }
        }
        // O += P @ V
#pragma unroll
        for (int vt = 0; vt < 4; ++vt) {
#pragma unroll
            for (int kk = 0; kk < 2; ++kk) {
                const f16* vb = &Vs[(kk * 32) * VSS + vt * 16 + lm];
                v16h bv;
#pragma unroll
                for (int h = 0; h < 16; ++h) bv[h] = vb[((h >> 3) * 16 + g * 8 + (h & 7)) * VSS];
                co[vt] = wmma_f16(ap[kk], bv, co[vt]);
            }
        }
    }

    // write attn output (T*B, E) f16:  row = t*B + b, col = h*64 + hd
    const int hh = bh & 15;
#pragma unroll
    for (int vt = 0; vt < 4; ++vt) {
        int e = hh * HDn + vt * 16 + lm;
#pragma unroll
        for (int j = 0; j < 8; ++j) {
            int t = q0 + g * 8 + j;
            float o = co[vt][j] / lrun[j];
            attnout[(size_t)(t * Bn + b) * En + e] = (f16)o;
        }
    }
}

// ---------------------------------------------------------------------------
// Kernel 3: output projection.  out = A(8192,1024 f16) @ Wout(1024,1024)^T + bout
// A tile staged via async global->LDS (ASYNCcnt) when available; W tile
// register-prefetched one k-step ahead.
// ---------------------------------------------------------------------------
__launch_bounds__(256)
__global__ void out_proj_kernel(const f16* __restrict__ A,
                                const float* __restrict__ Wout,
                                const float* __restrict__ bout,
                                float* __restrict__ out)
{
    constexpr int BM = 128, BN = 128, BK = 32;
    constexpr int AST = 40;   // 80B rows: 16B-aligned for b128 async stores,
                              // 20-word stride -> conflict-free 16-row gather
    constexpr int BST = 136;
    __shared__ f16 As[BM * AST];
    __shared__ f16 Bs[BK * BST];

    const int tid  = threadIdx.x;
    const int lane = tid & 31, wave = tid >> 5;
    const int wm = wave >> 1, wn = wave & 1;
    const int g = lane >> 4, lm = lane & 15;
    const int m0 = blockIdx.x * BM;
    const int n0 = blockIdx.y * BN;

    v8f zero8 = {0.f,0.f,0.f,0.f,0.f,0.f,0.f,0.f};
    v8f acc[2][4];
#pragma unroll
    for (int mi = 0; mi < 2; ++mi)
#pragma unroll
        for (int ni = 0; ni < 4; ++ni) acc[mi][ni] = zero8;

    const int arow = tid >> 1;
    const int ac16 = (tid & 1) * 16;
    const int wr = tid >> 3;
    const int wc = (tid & 7) * 4;

    float4 wT[4];
    auto load_w = [&](int k0) {
#pragma unroll
        for (int p = 0; p < 4; ++p)
            wT[p] = *(const float4*)&Wout[(size_t)(n0 + p * 32 + wr) * En + k0 + wc];
    };
    load_w(0);

    for (int k0 = 0; k0 < En; k0 += BK) {
        __syncthreads();
        // stage A tile (f16 copy, 16 halves per thread)
        {
            const f16* src = &A[(size_t)(m0 + arow) * En + k0 + ac16];
            f16* dst = &As[arow * AST + ac16];
#ifdef ASYNC_LDS
            __builtin_amdgcn_global_load_async_to_lds_b128((as1_i4p)src,       (as3_i4p)dst,       0, 0);
            __builtin_amdgcn_global_load_async_to_lds_b128((as1_i4p)(src + 8), (as3_i4p)(dst + 8), 0, 0);
#else
            v8h a0 = *(const v8h*)&src[0];
            v8h a1 = *(const v8h*)&src[8];
            *(v8h*)&dst[0] = a0;
            *(v8h*)&dst[8] = a1;
#endif
        }
        // stage W tile transposed from prefetched registers
#pragma unroll
        for (int p = 0; p < 4; ++p) {
            int n = p * 32 + wr;
            Bs[(wc + 0) * BST + n] = (f16)wT[p].x;
            Bs[(wc + 1) * BST + n] = (f16)wT[p].y;
            Bs[(wc + 2) * BST + n] = (f16)wT[p].z;
            Bs[(wc + 3) * BST + n] = (f16)wT[p].w;
        }
        if (k0 + BK < En) load_w(k0 + BK);   // in flight during WMMA below
#ifdef ASYNC_LDS
        __builtin_amdgcn_s_wait_asynccnt(0);
#endif
        __syncthreads();

        v16h af[2];
#pragma unroll
        for (int mi = 0; mi < 2; ++mi) {
            const f16* base = &As[(wm * 32 + mi * 16 + lm) * AST];
#pragma unroll
            for (int h = 0; h < 16; h += 2) {
                int k = (h >> 3) * 16 + g * 8 + (h & 7);
                h2 t = *(const h2*)&base[k];
                af[mi][h] = t[0]; af[mi][h + 1] = t[1];
            }
        }
#pragma unroll
        for (int ni = 0; ni < 4; ++ni) {
            const f16* bb = &Bs[wn * 64 + ni * 16 + lm];
            v16h bf;
#pragma unroll
            for (int h = 0; h < 16; ++h) bf[h] = bb[((h >> 3) * 16 + g * 8 + (h & 7)) * BST];
#pragma unroll
            for (int mi = 0; mi < 2; ++mi)
                acc[mi][ni] = wmma_f16(af[mi], bf, acc[mi][ni]);
        }
    }

#pragma unroll
    for (int ni = 0; ni < 4; ++ni) {
        int f = n0 + wn * 64 + ni * 16 + lm;
        float bias = bout[f];
#pragma unroll
        for (int mi = 0; mi < 2; ++mi) {
            int mbase = m0 + wm * 32 + mi * 16 + g * 8;
#pragma unroll
            for (int j = 0; j < 8; ++j) {
                int m = mbase + j;
                out[(size_t)m * En + f] = acc[mi][ni][j] + bias;
            }
        }
    }
}

// ---------------------------------------------------------------------------
extern "C" void kernel_launch(void* const* d_in, const int* in_sizes, int n_in,
                              void* d_out, int out_size, void* d_ws, size_t ws_size,
                              hipStream_t stream) {
    const float*         query = (const float*)d_in[0];
    const unsigned char* kpm   = (const unsigned char*)d_in[1];   // bool (B,T)
    const float*         mask  = (const float*)d_in[2];           // (T,T)
    const float*         Win   = (const float*)d_in[3];           // (3E,E)
    const float*         bin   = (const float*)d_in[4];           // (3E,)
    const float*         Wout  = (const float*)d_in[5];           // (E,E)
    const float*         bout  = (const float*)d_in[6];           // (E,)
    float* out = (float*)d_out;

    const size_t head_elems = (size_t)Bn * Hn * Tn * HDn;   // 8,388,608
    f16* Qb    = (f16*)d_ws;
    f16* Kb    = Qb + head_elems;
    f16* Vb    = Kb + head_elems;
    f16* attnb = Vb + head_elems;                           // (T*B, E) f16
    // total workspace: 4 * 16 MB = 64 MB

    qkv_proj_kernel<<<dim3(MROWS / 128, NQKV / 128), 256, 0, stream>>>(
        query, Win, bin, Qb, Kb, Vb);
    attn_kernel<<<dim3(Tn / 128, Bn * Hn), 256, 0, stream>>>(
        Qb, Kb, Vb, mask, kpm, attnb);
    out_proj_kernel<<<dim3(MROWS / 128, En / 128), 256, 0, stream>>>(
        attnb, Wout, bout, out);
}
